// Decoder_37426345017996
// MI455X (gfx1250) — matrix operand order repaired and checked by reading, not compiled, not run on hardware
//
#include <hip/hip_runtime.h>

// ---------------------------------------------------------------------------
// Problem constants (from reference): B=8192, L=128, H=256, A=32, T=65, E=8
// ---------------------------------------------------------------------------
#define BB   8192
#define LL   128
#define HH   256
#define AA   32
#define TT   65
#define EE   8

// hx tile: [h(256) | x(8) | zero-pad(24)]  -> K = 288 = 9*32
#define KAUG 288

typedef __attribute__((ext_vector_type(16))) __bf16       v16bf;
typedef __attribute__((ext_vector_type(8)))  float        v8f;
typedef __attribute__((ext_vector_type(4)))  unsigned int u32x4;
typedef __attribute__((ext_vector_type(8)))  int          i32x8;
typedef __attribute__((ext_vector_type(4)))  int          i32x4;

union FragU { v16bf v; u32x4 q[2]; };

// ---- packed-weight workspace layout (bf16 element offsets) ----------------
// B-fragment pack order inside a fragment: elem(lane*16 + c*8 + e)
//   n = tile*16 + (lane&15)
//   k = kt*32 + (lane>>4)*16 + c*8 + e        (contiguous k order, same as A loads)
//
// GATES region: 16 chunks (one per 16-col tile j), each 27 fragments:
//   f=0..8  : r-gate  [Whh|Wih] rows j*16..      kt=f
//   f=9..17 : z-gate  [Whh|Wih] rows 256+j*16..  kt=f-9
//   f=18..25: h_n     Whh rows 512+j*16..        kt=f-18
//   f=26    : i_n     Wih rows 512+j*16.. (K=32 padded)
#define GATE_CHUNK_ELEM  13824            // 27 frags * 512 bf16
#define GATE_CHUNK_BYTE  27648
#define OFF_G   0                         // 16 * 13824 = 221184
#define OFF_D0  221184                    // Wd0 [128x256]  16 tiles x 4 kt
#define OFF_D1  253952                    // Wd1 [256x256]  16 tiles x 8 kt
#define OFF_D2  319488                    // Wd2 [256x256]  16 tiles x 8 kt
#define OFF_P1  385024                    // Wp1 [256x32]    2 tiles x 8 kt
#define OFF_P2  393216                    // Wp2 [32x32]     2 tiles x 1 kt
#define PACK_TOTAL 394240
#define BC_BYTE_OFF 788480                // 4 x 256 combined-bias floats

// ---- LDS layout (single block, manual sub-allocation; byte offsets) -------
#define SM_HX    0                        // double-buffered hx: 2*64*288*2
#define SM_BLDS  73728                    // gate-weight double buffer: 2*27648
#define SM_HEADW 129024                   // Wp1/Wp2 fragments: 9216*2
#define SM_PLDS  147456                   // relu(h@Wp1) staging: 64*32*2
#define SM_BCL   151552                   // combined gate biases: 1024*4
#define SM_TOTAL 155648

static __device__ __forceinline__ void load_frag_lds(FragU& f, const __bf16* base,
                                                     int stride, int lane, int kt) {
    // A-operand layout: contiguous k = kt*32 + (lane>>4)*16 + s (matches B pack)
    int r  = lane & 15;
    int k0 = kt * 32 + ((lane >> 4) << 4);
    const __bf16* p = base + r * stride + k0;
    f.q[0] = *(const u32x4*)(p);
    f.q[1] = *(const u32x4*)(p + 8);
}

static __device__ __forceinline__ void load_frag_pk(FragU& f, const __bf16* pk,
                                                    int frag, int lane) {
    const __bf16* p = pk + ((size_t)frag * 32 + lane) * 16;
    f.q[0] = *(const u32x4*)(p);
    f.q[1] = *(const u32x4*)(p + 8);
}

static __device__ __forceinline__ v8f wmma_bf16(const FragU& a, const FragU& b, v8f c) {
    return __builtin_amdgcn_wmma_f32_16x16x32_bf16(
        false, a.v, false, b.v, (short)0, c, false, false);
}

static __device__ __forceinline__ float sigm(float x)   { return 1.0f / (1.0f + __expf(-x)); }
static __device__ __forceinline__ float tanh_f(float x) { return 2.0f / (1.0f + __expf(-2.0f * x)) - 1.0f; }

// 1-D Tensor Data Mover copy: global -> LDS, n8 elements of 8 bytes.
// D# per CDNA5 ISA: group0 {count=1, lds_addr, global_addr[56:0], type=2},
// group1 {data_size=3(8B), tensor_dim0=n8, tensor_dim1=1, tile_dim0=n8,
//         tensor_dim0_stride=n8}. Groups 2/3 unused (zeros).
// Toolchain here is the 6-arg flavor (clang-23/therock): extra int32x8 arg.
static __device__ __forceinline__ void tdm_load_1d(unsigned int lds_byte_off,
                                                   const void* gptr,
                                                   unsigned int n8) {
    unsigned long long ga = (unsigned long long)(size_t)gptr;
    u32x4 g0;
    g0[0] = 1u;                                       // count=1, user descriptor
    g0[1] = lds_byte_off;                             // lds_addr
    g0[2] = (unsigned int)(ga & 0xFFFFFFFFu);         // global_addr[31:0]
    g0[3] = (unsigned int)((ga >> 32) & 0x01FFFFFFu)  // global_addr[56:32]
          | (2u << 30);                               // type=2 ("image")
    i32x8 g1;
    g1[0] = (int)(3u << 16);                          // wg_mask=0, data_size=8B
    g1[1] = (int)((n8 & 0xFFFFu) << 16);              // tensor_dim0[15:0]
    g1[2] = (int)(((n8 >> 16) & 0xFFFFu) | (1u << 16)); // dim0[31:16] | tensor_dim1=1
    g1[3] = (int)((n8 & 0xFFFFu) << 16);              // tile_dim0
    g1[4] = 0;                                        // tile_dim1/2 unused
    g1[5] = (int)n8;                                  // tensor_dim0_stride[31:0]
    g1[6] = 0;
    g1[7] = 0;
    i32x4 z4 = {0, 0, 0, 0};
    i32x8 z8 = {0, 0, 0, 0, 0, 0, 0, 0};
    __builtin_amdgcn_tensor_load_to_lds(g0, g1, z4, z4, z8, 0);
}

// ---------------------------------------------------------------------------
// Phase 0: weights -> bf16 WMMA B-fragment order + fused bias table
// ---------------------------------------------------------------------------
__global__ __launch_bounds__(256) void pack_kernel(
    const float* __restrict__ W_ih, const float* __restrict__ b_ih,
    const float* __restrict__ W_hh, const float* __restrict__ b_hh,
    const float* __restrict__ Wd0,  const float* __restrict__ Wd1,
    const float* __restrict__ Wd2,  const float* __restrict__ Wp1,
    const float* __restrict__ Wp2,
    __bf16* __restrict__ pack, float* __restrict__ bc)
{
    int idx = blockIdx.x * 256 + threadIdx.x;
    if (idx >= PACK_TOTAL + 1024) return;

    if (idx >= PACK_TOTAL) {            // combined bias table bc[4][256]
        int i = idx - PACK_TOTAL;
        int g = i >> 8, n = i & 255;
        float v;
        if      (g == 0) v = b_ih[n]       + b_hh[n];        // r
        else if (g == 1) v = b_ih[256 + n] + b_hh[256 + n];  // z
        else if (g == 2) v = b_hh[512 + n];                  // h_n
        else             v = b_ih[512 + n];                  // i_n
        bc[i] = v;
        return;
    }

    if (idx < OFF_D0) {                 // GATES region (chunked per j)
        int j  = idx / GATE_CHUNK_ELEM;
        int r2 = idx % GATE_CHUNK_ELEM;
        int f  = r2 >> 9;               // fragment 0..26
        int w  = r2 & 511;
        int e = w & 7, cc = (w >> 3) & 1, lane = w >> 4;
        int nc = lane & 15, half = lane >> 4;
        float val = 0.0f;
        if (f < 18) {                                   // r (f<9) / z gates
            int kt = (f < 9) ? f : f - 9;
            int n  = ((f < 9) ? 0 : 256) + j * 16 + nc;
            int k  = kt * 32 + half * 16 + cc * 8 + e;
            val = (k < 256) ? W_hh[n * 256 + k]
                : ((k < 256 + EE) ? W_ih[n * EE + (k - 256)] : 0.0f);
        } else if (f < 26) {                            // h_n
            int kt = f - 18;
            int n  = 512 + j * 16 + nc;
            int k  = kt * 32 + half * 16 + cc * 8 + e;
            val = W_hh[n * 256 + k];
        } else {                                        // i_n (x cols, padded)
            int n = 512 + j * 16 + nc;
            int k = half * 16 + cc * 8 + e;
            val = (k < EE) ? W_ih[n * EE + k] : 0.0f;
        }
        pack[idx] = (__bf16)val;
        return;
    }

    int region, rel, KT;
    if      (idx < OFF_D1) { region = 3; rel = idx - OFF_D0; KT = 4; }
    else if (idx < OFF_D2) { region = 4; rel = idx - OFF_D1; KT = 8; }
    else if (idx < OFF_P1) { region = 5; rel = idx - OFF_D2; KT = 8; }
    else if (idx < OFF_P2) { region = 6; rel = idx - OFF_P1; KT = 8; }
    else                   { region = 7; rel = idx - OFF_P2; KT = 1; }

    int e     = rel & 7;
    int c     = (rel >> 3) & 1;
    int lane  = (rel >> 4) & 31;
    int ktn   = rel >> 9;
    int kt    = ktn % KT;
    int tileN = ktn / KT;
    int n = tileN * 16 + (lane & 15);
    int k = kt * 32 + ((lane >> 4) << 4) + c * 8 + e;

    float val = 0.0f;
    switch (region) {
        case 3: val = Wd0[k * 256 + n]; break;
        case 4: val = Wd1[k * 256 + n]; break;
        case 5: val = Wd2[k * 256 + n]; break;
        case 6: val = Wp1[k * 32 + n];  break;
        case 7: val = Wp2[k * 32 + n];  break;
    }
    pack[idx] = (__bf16)val;
}

// ---------------------------------------------------------------------------
// Dense 16x16-tiled GEMM into LDS (init-only MLP head; B frags from global)
// ---------------------------------------------------------------------------
static __device__ __forceinline__ void dense_gemm(
    const __bf16* src, __bf16* dst, const __bf16* packW, const float* bias,
    int KT, int doRelu, int row0, int lane)
{
    const int nC = lane & 15;
    const int m0 = (lane >> 4) << 3;
    for (int j = 0; j < 16; ++j) {
        v8f acc = {0,0,0,0,0,0,0,0};
        FragU a, b;
        for (int kt = 0; kt < KT; ++kt) {
            load_frag_lds(a, src + row0 * KAUG, KAUG, lane, kt);
            load_frag_pk(b, packW, j * KT + kt, lane);
            acc = wmma_bf16(a, b, acc);
        }
        int col = j * 16 + nC;
        float bv = bias[col];
        #pragma unroll
        for (int v = 0; v < 8; ++v) {
            float x = acc[v] + bv;
            if (doRelu) x = x > 0.0f ? x : 0.0f;
            dst[(row0 + m0 + v) * KAUG + col] = (__bf16)x;
        }
    }
}

// ---------------------------------------------------------------------------
// Persistent fused decoder. 128 blocks x 128 threads (4 wave32); each wave
// owns 16 batch rows. Gate weights are streamed global->LDS with the TDM
// (double-buffered, one chunk per 16-col tile), so every weight byte crosses
// the L2->WGP link once per block instead of once per wave.
// ---------------------------------------------------------------------------
__global__ __launch_bounds__(128) void decoder_kernel(
    const float* __restrict__ latent, const int* __restrict__ target,
    const float* __restrict__ embed,
    const float* __restrict__ bd0, const float* __restrict__ bd1,
    const float* __restrict__ bd2, const float* __restrict__ bp1,
    const float* __restrict__ bp2,
    const __bf16* __restrict__ pack, const float* __restrict__ bc,
    float* __restrict__ out)
{
    __shared__ __attribute__((aligned(128))) unsigned char smem[SM_TOTAL];
    __bf16* hxs   = (__bf16*)(smem + SM_HX);       // [2][64*288]
    __bf16* blds  = (__bf16*)(smem + SM_BLDS);     // [2][13824]
    __bf16* headw = (__bf16*)(smem + SM_HEADW);    // P1 frags + P2 frags
    __bf16* plds  = (__bf16*)(smem + SM_PLDS);     // [64*32]
    float*  bcl   = (float*)(smem + SM_BCL);       // [1024]

    const int tid  = threadIdx.x;
    const int lane = tid & 31;
    const int wave = tid >> 5;
    const int row0 = wave * 16;
    const int b0   = blockIdx.x * 64;
    const int nC   = lane & 15;
    const int m0   = (lane >> 4) << 3;

    // ---- init --------------------------------------------------------------
    {
        unsigned int* z = (unsigned int*)smem;
        for (int i = tid; i < SM_TOTAL / 4; i += 128) z[i] = 0u;
    }
    __syncthreads();

    for (int i = tid; i < 1024; i += 128) bcl[i] = bc[i];
    for (int i = tid; i < 9216; i += 128) headw[i] = pack[OFF_P1 + i];

    // latent tile -> hx buffer 1 (cols 0..127); x0 = embed[0] -> buffer 0
    for (int i = lane; i < 16 * LL; i += 32) {
        int r = i >> 7, c = i & (LL - 1);
        hxs[64 * KAUG + (row0 + r) * KAUG + c] =
            (__bf16)latent[(size_t)(b0 + row0 + r) * LL + c];
    }
    for (int i = lane; i < 16 * EE; i += 32) {
        int r = i >> 3, e = i & (EE - 1);
        hxs[(row0 + r) * KAUG + 256 + e] = (__bf16)embed[e];
    }
    __syncthreads();

    float bp1v0 = bp1[nC], bp1v1 = bp1[16 + nC];
    float bp2v0 = bp2[nC], bp2v1 = bp2[16 + nC];

    // Prefetch gate chunk j=0 into LDS buffer 0 while the MLP head runs.
    if (wave == 0)
        tdm_load_1d(SM_BLDS, (const char*)pack, GATE_CHUNK_BYTE / 8);

    // ---- MLP head: h0 = relu(relu(latent@Wd0+bd0)@Wd1+bd1)@Wd2+bd2 ---------
    dense_gemm(hxs + 64 * KAUG, hxs,             pack + OFF_D0, bd0, 4, 1, row0, lane);
    dense_gemm(hxs,             hxs + 64 * KAUG, pack + OFF_D1, bd1, 8, 1, row0, lane);
    dense_gemm(hxs + 64 * KAUG, hxs,             pack + OFF_D2, bd2, 8, 0, row0, lane);
    // hx buffer 0 now holds [h0 | x0 | 0]

    // ---- time loop ---------------------------------------------------------
    for (int t = 0; t < TT; ++t) {
        const int cur = t & 1;
        __bf16* hxc = hxs + cur * (64 * KAUG);
        __bf16* hxn = hxs + (cur ^ 1) * (64 * KAUG);

        // Preload all 9 A-fragments of this wave's [h|x|pad] strip.
        FragU Af[9];
        #pragma unroll
        for (int kt = 0; kt < 9; ++kt)
            load_frag_lds(Af[kt], hxc + row0 * KAUG, KAUG, lane, kt);

        for (int j = 0; j < 16; ++j) {
            // chunk j is in buffer (j&1); wait for it, then start chunk j+1
            // into the other buffer (safe: barrier proves everyone is done
            // reading that buffer from the previous iteration).
            if (wave == 0) __builtin_amdgcn_s_wait_tensorcnt((short)0);
            __syncthreads();
            if (wave == 0)
                tdm_load_1d(SM_BLDS + (unsigned)(((j + 1) & 1) * GATE_CHUNK_BYTE),
                            (const char*)pack + (size_t)((j + 1) & 15) * GATE_CHUNK_BYTE,
                            GATE_CHUNK_BYTE / 8);
            const __bf16* Bl = blds + (j & 1) * GATE_CHUNK_ELEM;

            v8f ar = {0,0,0,0,0,0,0,0};
            v8f az = ar, ahn = ar, ain = ar;
            FragU bf;
            #pragma unroll
            for (int kt = 0; kt < 9; ++kt) {
                load_frag_pk(bf, Bl, kt, lane);            // r gate
                ar = wmma_bf16(Af[kt], bf, ar);
                load_frag_pk(bf, Bl, 9 + kt, lane);        // z gate
                az = wmma_bf16(Af[kt], bf, az);
                if (kt < 8) {
                    load_frag_pk(bf, Bl, 18 + kt, lane);   // h_n
                    ahn = wmma_bf16(Af[kt], bf, ahn);
                }
            }
            load_frag_pk(bf, Bl, 26, lane);                // i_n (x columns)
            ain = wmma_bf16(Af[8], bf, ain);

            int col = j * 16 + nC;
            float bcr  = bcl[col];
            float bcz  = bcl[256 + col];
            float bchn = bcl[512 + col];
            float bcin = bcl[768 + col];
            #pragma unroll
            for (int v = 0; v < 8; ++v) {
                float hprev = (float)hxc[(row0 + m0 + v) * KAUG + col];
                float r  = sigm(ar[v] + bcr);
                float z  = sigm(az[v] + bcz);
                float nn = tanh_f(ain[v] + bcin + r * (ahn[v] + bchn));
                float hnew = (1.0f - z) * nn + z * hprev;
                hxn[(row0 + m0 + v) * KAUG + col] = (__bf16)hnew;
            }
        }

        // ---- output head: logits = relu(h_new@Wp1+bp1)@Wp2+bp2 -------------
        FragU HA[8];
        #pragma unroll
        for (int kt = 0; kt < 8; ++kt)
            load_frag_lds(HA[kt], hxn + row0 * KAUG, KAUG, lane, kt);

        #pragma unroll
        for (int a = 0; a < 2; ++a) {
            v8f acc = {0,0,0,0,0,0,0,0};
            FragU bf;
            #pragma unroll
            for (int kt = 0; kt < 8; ++kt) {
                load_frag_pk(bf, headw, a * 8 + kt, lane);     // Wp1 (LDS)
                acc = wmma_bf16(HA[kt], bf, acc);
            }
            float bb = a ? bp1v1 : bp1v0;
            #pragma unroll
            for (int v = 0; v < 8; ++v) {
                float x = acc[v] + bb;
                x = x > 0.0f ? x : 0.0f;
                plds[(row0 + m0 + v) * 32 + a * 16 + nC] = (__bf16)x;
            }
        }

        FragU PA;
        load_frag_lds(PA, plds + row0 * 32, 32, lane, 0);
        #pragma unroll
        for (int a = 0; a < 2; ++a) {
            v8f acc = {0,0,0,0,0,0,0,0};
            FragU bf;
            load_frag_pk(bf, headw + 8192, a, lane);           // Wp2 (LDS)
            acc = wmma_bf16(PA, bf, acc);
            float bb = a ? bp2v1 : bp2v0;
            #pragma unroll
            for (int v = 0; v < 8; ++v) {
                out[(size_t)(b0 + row0 + m0 + v) * (TT * AA) + t * AA + a * 16 + nC] =
                    acc[v] + bb;
            }
        }

        // ---- stage x_{t+1} = embed[target[:, t]] into next buffer ----------
        if (t + 1 < TT) {
            int rr = row0 + (lane >> 1);
            int e0 = (lane & 1) * 4;
            int tok = target[(size_t)(b0 + rr) * TT + t];
            #pragma unroll
            for (int e = 0; e < 4; ++e)
                hxn[rr * KAUG + 256 + e0 + e] = (__bf16)embed[tok * EE + e0 + e];
        }
    }
}

// ---------------------------------------------------------------------------
extern "C" void kernel_launch(void* const* d_in, const int* in_sizes, int n_in,
                              void* d_out, int out_size, void* d_ws, size_t ws_size,
                              hipStream_t stream) {
    (void)in_sizes; (void)n_in; (void)out_size; (void)ws_size;
    const float* latent = (const float*)d_in[0];
    const int*   target = (const int*)  d_in[1];
    const float* embed  = (const float*)d_in[2];
    const float* W_ih   = (const float*)d_in[3];
    const float* b_ih   = (const float*)d_in[4];
    const float* W_hh   = (const float*)d_in[5];
    const float* b_hh   = (const float*)d_in[6];
    const float* Wd0    = (const float*)d_in[7];
    const float* bd0    = (const float*)d_in[8];
    const float* Wd1    = (const float*)d_in[9];
    const float* bd1    = (const float*)d_in[10];
    const float* Wd2    = (const float*)d_in[11];
    const float* bd2    = (const float*)d_in[12];
    const float* Wp1    = (const float*)d_in[13];
    const float* bp1    = (const float*)d_in[14];
    const float* Wp2    = (const float*)d_in[15];
    const float* bp2    = (const float*)d_in[16];
    float* out = (float*)d_out;

    __bf16* pack = (__bf16*)d_ws;
    float*  bc   = (float*)((char*)d_ws + BC_BYTE_OFF);

    const int packThreads = PACK_TOTAL + 1024;
    pack_kernel<<<(packThreads + 255) / 256, 256, 0, stream>>>(
        W_ih, b_ih, W_hh, b_hh, Wd0, Wd1, Wd2, Wp1, Wp2, pack, bc);

    decoder_kernel<<<BB / 64, 128, 0, stream>>>(
        latent, target, embed, bd0, bd1, bd2, bp1, bp2, pack, bc, out);
}
